// raw_aux_siamese_8048768713004
// MI455X (gfx1250) — matrix operand order repaired and checked by reading, not compiled, not run on hardware
//
#include <hip/hip_runtime.h>

typedef __attribute__((ext_vector_type(16))) _Float16 v16h;
typedef __attribute__((ext_vector_type(8)))  float    v8f;

#define DEVINL __device__ __forceinline__

DEVINL float sigm(float x)   { return 1.0f / (1.0f + __expf(-x)); }
DEVINL float tanh_f(float x) { return 1.0f - 2.0f / (__expf(2.0f * x) + 1.0f); }

// ---------------------------------------------------------------------------
// Kernel 1: 50 directional LSTM chains (25 encoders x 2 dirs), T=200, B=512,
// H=32, input size 1.  One wave per (chain, 16 batch rows).
//
// Recurrence per step, per gate tile j (gates permuted G = q*32 + r + 8p + 16hi):
//   D2 = wmma(A2_j, B2, 0)       // A2: K=0 -> Wih, K=1 -> bias;  B2: K=0 -> x_b, K=1 -> 1
//   D  = wmma(A_j,  hB, D2)      // Whh(permuted) x h^T + input projection + bias
// The gate permutation makes the D layout identical to the next step's B
// layout (lane-half hi holds hidden k = 16hi..16hi+15), so h feeds back with
// zero cross-lane traffic.
//
// chain = 2e + dir; e<5: encA on x1 mark e; e<10: encA on x2 mark e-5;
// e>=10: encX on xc mark e-10 (xc = [x1, x2, x1-x2]).
// h_seq (f16): [50][200][512][32]
// ---------------------------------------------------------------------------
__global__ __launch_bounds__(128)
void lstm_marks_kernel(const float* __restrict__ x1, const float* __restrict__ x2,
                       const float* __restrict__ encA_Wih, const float* __restrict__ encA_Whh,
                       const float* __restrict__ encA_b,
                       const float* __restrict__ encX_Wih, const float* __restrict__ encX_Whh,
                       const float* __restrict__ encX_b,
                       _Float16* __restrict__ h_seq)
{
  const int T = 200, B = 512;
  const int chain = blockIdx.x;      // 0..49
  const int wg    = blockIdx.y;      // 0..7  (64 batch rows each)
  const int e = chain >> 1, dir = chain & 1;
  const int stream = (e < 5) ? 0 : (e < 10) ? 1 : 2;
  const int mark   = (e < 5) ? e : (e < 10) ? (e - 5) : (e - 10);

  const float *Wih, *Whh, *bias;
  if (stream < 2) {
    int wi = mark * 2 + dir;
    Wih  = encA_Wih + wi * 128;
    Whh  = encA_Whh + wi * 128 * 32;
    bias = encA_b   + wi * 128;
  } else {
    int wi = mark * 2 + dir;
    Wih  = encX_Wih + wi * 128;
    Whh  = encX_Whh + wi * 128 * 32;
    bias = encX_b   + wi * 128;
  }

  __shared__ float xs[64][200];
  const int tid = threadIdx.x;
  for (int idx = tid; idx < 64 * 200; idx += 128) {
    int r = idx / 200, t = idx % 200;
    int b = wg * 64 + r;
    float v;
    if (stream == 0)      v = x1[(b * 200 + t) * 5 + mark];
    else if (stream == 1) v = x2[(b * 200 + t) * 5 + mark];
    else {
      int mm = mark;
      if (mm < 5)       v = x1[(b * 200 + t) * 5 + mm];
      else if (mm < 10) v = x2[(b * 200 + t) * 5 + mm - 5];
      else              v = x1[(b * 200 + t) * 5 + mm - 10] - x2[(b * 200 + t) * 5 + mm - 10];
    }
    xs[r][t] = v;
  }
  __syncthreads();

  const int wv = tid >> 5, lane = tid & 31;
  const int bl = lane & 15, hi = lane >> 4;
  const int rloc  = wv * 16 + bl;     // batch row within block
  const int bglob = wg * 64 + rloc;   // global batch row (this lane's N column)

  // A tiles: gate-permuted Whh rows as WMMA A operands, contiguous k order (same as hB):
  // lane row M = lane%16; half u -> K = 16hi+u.
  v16h A[8];
  #pragma unroll
  for (int j = 0; j < 8; ++j) {
    const int q = j & 3, p = j >> 2;
    const int G = q * 32 + (bl & 7) + 8 * p + 16 * (bl >> 3);
    const float* wr = Whh + G * 32;
    #pragma unroll
    for (int u = 0; u < 8; ++u) A[j][u]     = (_Float16)wr[16 * hi + u];
    #pragma unroll
    for (int u = 0; u < 8; ++u) A[j][8 + u] = (_Float16)wr[16 * hi + 8 + u];
  }

  // A2 tiles: rank-1 input projection + bias.  Row M = bl, K=0 -> Wih[G],
  // K=1 -> bias[G].  K=0,1 halves live in lane-group hi==0 only.
  v16h A2[8];
  #pragma unroll
  for (int j = 0; j < 8; ++j) {
    const int q = j & 3, p = j >> 2;
    const int G = q * 32 + (bl & 7) + 8 * p + 16 * (bl >> 3);
    #pragma unroll
    for (int u = 0; u < 16; ++u) A2[j][u] = (_Float16)0.0f;
    A2[j][0] = (hi == 0) ? (_Float16)Wih[G]  : (_Float16)0.0f;
    A2[j][1] = (hi == 0) ? (_Float16)bias[G] : (_Float16)0.0f;
  }

  // B2: column N = bl; K=0 -> x (updated per step), K=1 -> 1.0 (invariant).
  v16h B2;
  #pragma unroll
  for (int i = 0; i < 16; ++i) B2[i] = (_Float16)0.0f;
  B2[1] = (hi == 0) ? (_Float16)1.0f : (_Float16)0.0f;

  float cst[16];
  #pragma unroll
  for (int i = 0; i < 16; ++i) cst[i] = 0.0f;
  v16h hB;                       // B operand: lane = batch col bl, K = 16hi + half
  #pragma unroll
  for (int i = 0; i < 16; ++i) hB[i] = (_Float16)0.0f;
  const v8f zero8 = {};

  for (int step = 0; step < T; ++step) {
    const int t = dir ? (T - 1 - step) : step;
    const float xv = xs[rloc][t];
    B2[0] = (hi == 0) ? (_Float16)xv : (_Float16)0.0f;
    _Float16 hnew[16];
    #pragma unroll
    for (int p = 0; p < 2; ++p) {
      v8f Pi = __builtin_amdgcn_wmma_f32_16x16x32_f16(false, A2[0 + 4 * p], false, B2, (short)0, zero8, false, false);
      v8f Pf = __builtin_amdgcn_wmma_f32_16x16x32_f16(false, A2[1 + 4 * p], false, B2, (short)0, zero8, false, false);
      v8f Pg = __builtin_amdgcn_wmma_f32_16x16x32_f16(false, A2[2 + 4 * p], false, B2, (short)0, zero8, false, false);
      v8f Po = __builtin_amdgcn_wmma_f32_16x16x32_f16(false, A2[3 + 4 * p], false, B2, (short)0, zero8, false, false);
      v8f Di = __builtin_amdgcn_wmma_f32_16x16x32_f16(false, A[0 + 4 * p], false, hB, (short)0, Pi, false, false);
      v8f Df = __builtin_amdgcn_wmma_f32_16x16x32_f16(false, A[1 + 4 * p], false, hB, (short)0, Pf, false, false);
      v8f Dg = __builtin_amdgcn_wmma_f32_16x16x32_f16(false, A[2 + 4 * p], false, hB, (short)0, Pg, false, false);
      v8f Do_= __builtin_amdgcn_wmma_f32_16x16x32_f16(false, A[3 + 4 * p], false, hB, (short)0, Po, false, false);
      #pragma unroll
      for (int r = 0; r < 8; ++r) {
        float cn = sigm(Df[r]) * cst[p * 8 + r] + sigm(Di[r]) * tanh_f(Dg[r]);
        cst[p * 8 + r] = cn;
        hnew[p * 8 + r] = (_Float16)(sigm(Do_[r]) * tanh_f(cn));   // hidden k = 16hi + 8p + r
      }
    }
    #pragma unroll
    for (int i = 0; i < 16; ++i) hB[i] = hnew[i];
    // store h: lane's 16 halves are contiguous hidden k = 16hi..16hi+15
    _Float16* hp = h_seq + (((size_t)chain * T + t) * B + bglob) * 32 + 16 * hi;
    const float4* hv4 = (const float4*)&hB;
    *(float4*)(hp)     = hv4[0];
    *(float4*)(hp + 8) = hv4[1];
  }
}

// ---------------------------------------------------------------------------
// Kernel 2: per-mark attention.  score[b,t] = fwd_h.ctx[0:32] + bwd_h.ctx[32:64],
// online softmax pass + weighted-sum pass.  Writes reps[25][512][64] (ws),
// bin_ax (encX alphas) and alpha_rep_1/2 (reps copies) into d_out.
// ---------------------------------------------------------------------------
__global__ __launch_bounds__(256)
void mark_attn_kernel(const _Float16* __restrict__ h_seq,
                      const float* __restrict__ encA_ctx, const float* __restrict__ encX_ctx,
                      float* __restrict__ reps, float* __restrict__ out)
{
  const int T = 200, B = 512;
  const int e = blockIdx.x;                        // 0..24
  const int b = blockIdx.y * 256 + threadIdx.x;    // 0..511
  const float* ctx = (e < 5)  ? (encA_ctx + e * 64)
                   : (e < 10) ? (encA_ctx + (e - 5) * 64)
                              : (encX_ctx + (e - 10) * 64);
  float ctxr[64];
  #pragma unroll
  for (int k = 0; k < 64; ++k) ctxr[k] = ctx[k];

  const _Float16* fbase = h_seq + ((size_t)(2 * e)     * T) * B * 32 + (size_t)b * 32;
  const _Float16* bbase = h_seq + ((size_t)(2 * e + 1) * T) * B * 32 + (size_t)b * 32;

  float mmax = -3.0e38f, ssum = 0.0f;
  for (int t = 0; t < T; ++t) {
    const _Float16* fp = fbase + (size_t)t * B * 32;
    const _Float16* bp = bbase + (size_t)t * B * 32;
    float sc = 0.0f;
    #pragma unroll
    for (int k = 0; k < 32; ++k) sc += (float)fp[k] * ctxr[k];
    #pragma unroll
    for (int k = 0; k < 32; ++k) sc += (float)bp[k] * ctxr[32 + k];
    float nm = fmaxf(mmax, sc);
    ssum = ssum * __expf(mmax - nm) + __expf(sc - nm);
    mmax = nm;
  }
  const float inv = 1.0f / ssum;

  float rep[64];
  #pragma unroll
  for (int k = 0; k < 64; ++k) rep[k] = 0.0f;
  for (int t = 0; t < T; ++t) {
    const _Float16* fp = fbase + (size_t)t * B * 32;
    const _Float16* bp = bbase + (size_t)t * B * 32;
    float sc = 0.0f;
    #pragma unroll
    for (int k = 0; k < 32; ++k) sc += (float)fp[k] * ctxr[k];
    #pragma unroll
    for (int k = 0; k < 32; ++k) sc += (float)bp[k] * ctxr[32 + k];
    const float a = __expf(sc - mmax) * inv;
    #pragma unroll
    for (int k = 0; k < 32; ++k) rep[k]      += a * (float)fp[k];
    #pragma unroll
    for (int k = 0; k < 32; ++k) rep[32 + k] += a * (float)bp[k];
    if (e >= 10) out[13312 + (size_t)b * 3000 + (e - 10) * 200 + t] = a;  // bin_ax
  }
  float* rp = reps + ((size_t)e * B + b) * 64;
  #pragma unroll
  for (int k = 0; k < 64; ++k) rp[k] = rep[k];
  if (e < 5) {
    float* o = out + 1549312 + (size_t)b * 320 + e * 64;        // alpha_rep_1
    #pragma unroll
    for (int k = 0; k < 64; ++k) o[k] = rep[k];
  } else if (e < 10) {
    float* o = out + 1713152 + (size_t)b * 320 + (e - 5) * 64;  // alpha_rep_2
    #pragma unroll
    for (int k = 0; k < 64; ++k) o[k] = rep[k];
  }
}

// ---------------------------------------------------------------------------
// Kernel 3: hm-level LSTMs (3 encoders x 2 dirs), seq over marks (L=5,5,25),
// input 64, hidden 16.  <0.5% of FLOPs -> scalar per-batch-row threads with
// LDS-staged weights.  hm_h (f32): [6][25][512][16].
// ---------------------------------------------------------------------------
__global__ __launch_bounds__(256)
void hm_lstm_kernel(const float* __restrict__ reps,
                    const float* __restrict__ hm_Wih, const float* __restrict__ hm_Whh,
                    const float* __restrict__ hm_b, float* __restrict__ hm_h)
{
  const int B = 512;
  const int hc = blockIdx.x;               // 0..5
  const int henc = hc >> 1, dir = hc & 1;
  const int L = (henc < 2) ? 5 : 25;
  const int b = blockIdx.y * 256 + threadIdx.x;

  __shared__ float wih_s[64 * 64];
  __shared__ float whh_s[64 * 16];
  __shared__ float bia_s[64];
  const float* gw = hm_Wih + hc * 64 * 64;
  const float* gu = hm_Whh + hc * 64 * 16;
  const float* gb = hm_b + hc * 64;
  for (int i = threadIdx.x; i < 64 * 64; i += 256) wih_s[i] = gw[i];
  for (int i = threadIdx.x; i < 64 * 16; i += 256) whh_s[i] = gu[i];
  for (int i = threadIdx.x; i < 64;      i += 256) bia_s[i] = gb[i];
  __syncthreads();

  float h[16], c[16];
  #pragma unroll
  for (int k = 0; k < 16; ++k) { h[k] = 0.0f; c[k] = 0.0f; }

  for (int step = 0; step < L; ++step) {
    const int t = dir ? (L - 1 - step) : step;
    const int enc = (henc == 0) ? t : (henc == 1) ? (5 + t)
                  : ((t < 15) ? (10 + t) : (t - 15));   // rep3 = [repsx, reps1, reps2]
    const float4* in4 = (const float4*)(reps + ((size_t)enc * B + b) * 64);
    float inr[64];
    #pragma unroll
    for (int k4 = 0; k4 < 16; ++k4) {
      float4 v = in4[k4];
      inr[4 * k4] = v.x; inr[4 * k4 + 1] = v.y; inr[4 * k4 + 2] = v.z; inr[4 * k4 + 3] = v.w;
    }
    float g[64];
    #pragma unroll
    for (int G = 0; G < 64; ++G) {
      float acc = bia_s[G];
      const float4* w4 = (const float4*)(wih_s + G * 64);
      #pragma unroll
      for (int k4 = 0; k4 < 16; ++k4) {
        float4 w = w4[k4];
        acc += w.x * inr[4 * k4] + w.y * inr[4 * k4 + 1] + w.z * inr[4 * k4 + 2] + w.w * inr[4 * k4 + 3];
      }
      const float4* u4 = (const float4*)(whh_s + G * 16);
      #pragma unroll
      for (int k4 = 0; k4 < 4; ++k4) {
        float4 u = u4[k4];
        acc += u.x * h[4 * k4] + u.y * h[4 * k4 + 1] + u.z * h[4 * k4 + 2] + u.w * h[4 * k4 + 3];
      }
      g[G] = acc;
    }
    float hn[16];
    #pragma unroll
    for (int k = 0; k < 16; ++k) {
      float cn = sigm(g[16 + k]) * c[k] + sigm(g[k]) * tanh_f(g[32 + k]);
      c[k] = cn;
      hn[k] = sigm(g[48 + k]) * tanh_f(cn);
    }
    #pragma unroll
    for (int k = 0; k < 16; ++k) h[k] = hn[k];
    float* hp = hm_h + (((size_t)hc * 25 + t) * B + b) * 16;
    #pragma unroll
    for (int k = 0; k < 16; ++k) hp[k] = h[k];
  }
}

// ---------------------------------------------------------------------------
// Kernel 4: hm-level attention + linear heads.  Writes attn1x and the three
// prediction vectors into d_out.
// ---------------------------------------------------------------------------
__global__ __launch_bounds__(256)
void hm_attn_head_kernel(const float* __restrict__ hm_h,
                         const float* __restrict__ hm_ctx,
                         const float* __restrict__ head_W, const float* __restrict__ head_b,
                         float* __restrict__ out)
{
  const int B = 512;
  const int henc = blockIdx.x;                       // 0..2
  const int b = blockIdx.y * 256 + threadIdx.x;
  const int L = (henc < 2) ? 5 : 25;
  float ctxr[32];
  #pragma unroll
  for (int k = 0; k < 32; ++k) ctxr[k] = hm_ctx[henc * 32 + k];

  const float* fb = hm_h + ((size_t)(2 * henc)     * 25) * B * 16 + (size_t)b * 16;
  const float* bb = hm_h + ((size_t)(2 * henc + 1) * 25) * B * 16 + (size_t)b * 16;

  float sc[25];
  float mmax = -3.0e38f;
  #pragma unroll
  for (int t = 0; t < 25; ++t) {
    if (t < L) {
      const float* fp = fb + (size_t)t * B * 16;
      const float* bp = bb + (size_t)t * B * 16;
      float s = 0.0f;
      #pragma unroll
      for (int k = 0; k < 16; ++k) s += fp[k] * ctxr[k] + bp[k] * ctxr[16 + k];
      sc[t] = s;
      mmax = fmaxf(mmax, s);
    } else {
      sc[t] = -3.0e38f;
    }
  }
  float ssum = 0.0f;
  #pragma unroll
  for (int t = 0; t < 25; ++t) ssum += __expf(sc[t] - mmax);
  const float inv = 1.0f / ssum;

  float rep[32];
  #pragma unroll
  for (int k = 0; k < 32; ++k) rep[k] = 0.0f;
  #pragma unroll
  for (int t = 0; t < 25; ++t) {
    if (t < L) {
      const float* fp = fb + (size_t)t * B * 16;
      const float* bp = bb + (size_t)t * B * 16;
      const float a = __expf(sc[t] - mmax) * inv;
      #pragma unroll
      for (int k = 0; k < 16; ++k) { rep[k] += a * fp[k]; rep[16 + k] += a * bp[k]; }
      if (henc == 2) out[512 + (size_t)b * 25 + t] = a;   // attn1x
    }
  }
  float pred = head_b[henc];
  #pragma unroll
  for (int k = 0; k < 32; ++k) pred += rep[k] * head_W[henc * 32 + k];
  if (henc == 0)      out[1876992 + b] = pred;   // prediction1
  else if (henc == 1) out[1877504 + b] = pred;   // prediction2
  else                out[b]           = pred;   // prediction
}

// ---------------------------------------------------------------------------
// Workspace layout (requires ws_size >= ~336 MB):
//   h_seq f16 [50][200][512][32] : 327,680,000 B @ 0
//   reps  f32 [25][512][64]      :   3,276,800 B @ 327,680,000
//   hm_h  f32 [6][25][512][16]   :   4,915,200 B @ 330,956,800
// ---------------------------------------------------------------------------
extern "C" void kernel_launch(void* const* d_in, const int* in_sizes, int n_in,
                              void* d_out, int out_size, void* d_ws, size_t ws_size,
                              hipStream_t stream) {
  const float* x1       = (const float*)d_in[0];
  const float* x2       = (const float*)d_in[1];
  const float* encA_Wih = (const float*)d_in[2];
  const float* encA_Whh = (const float*)d_in[3];
  const float* encA_b   = (const float*)d_in[4];
  const float* encA_ctx = (const float*)d_in[5];
  const float* encX_Wih = (const float*)d_in[6];
  const float* encX_Whh = (const float*)d_in[7];
  const float* encX_b   = (const float*)d_in[8];
  const float* encX_ctx = (const float*)d_in[9];
  const float* hm_Wih   = (const float*)d_in[10];
  const float* hm_Whh   = (const float*)d_in[11];
  const float* hm_b     = (const float*)d_in[12];
  const float* hm_ctx   = (const float*)d_in[13];
  const float* head_W   = (const float*)d_in[14];
  const float* head_b   = (const float*)d_in[15];
  float* out = (float*)d_out;

  char* ws = (char*)d_ws;
  _Float16* h_seq = (_Float16*)ws;
  float*    reps  = (float*)(ws + 327680000);
  float*    hm_h  = (float*)(ws + 327680000 + 3276800);

  lstm_marks_kernel<<<dim3(50, 8), 128, 0, stream>>>(x1, x2, encA_Wih, encA_Whh, encA_b,
                                                     encX_Wih, encX_Whh, encX_b, h_seq);
  mark_attn_kernel<<<dim3(25, 2), 256, 0, stream>>>(h_seq, encA_ctx, encX_ctx, reps, out);
  hm_lstm_kernel<<<dim3(6, 2), 256, 0, stream>>>(reps, hm_Wih, hm_Whh, hm_b, hm_h);
  hm_attn_head_kernel<<<dim3(3, 2), 256, 0, stream>>>(hm_h, hm_ctx, head_W, head_b, out);
}
